// TNDMPNNStep_88811333747467
// MI455X (gfx1250) — matrix operand order repaired and checked by reading, not compiled, not run on hardware
//
#include <hip/hip_runtime.h>
#include <hip/hip_bf16.h>

// ---------------------------------------------------------------------------
// MPNN step for MI455X (gfx1250, wave32, WMMA).
//   messages = relu(concat(nodes[s], nodes[r], edges) @ W_edge + b_edge)
//   recv     = segment_sum(messages, receivers)
//   new_nodes= GRU(carry=nodes, inputs=recv)
// bf16 WMMA (f32 accumulate) for all GEMMs; f32 gate math.
// f32->bf16 pair conversion = 2x v_add_u32 + 1x v_perm_b32 (round-to-nearest).
// Edge kernel software-pipelines A-tile loads across the WMMA block.
// NT hints keep the 192MB L2 for the nodes gather table / recv atomics.
// ---------------------------------------------------------------------------

#define DIM      128
#define NNODES   20000
#define NEDGES   640000

typedef __attribute__((ext_vector_type(16))) __bf16 v16bf;
typedef __attribute__((ext_vector_type(8)))  float  v8f;
typedef __attribute__((ext_vector_type(4)))  float  f4;

union AFrag { unsigned int ui[8]; v16bf v; };

// f32 pair -> packed bf16 (round-to-nearest, ties-away): 3 full-rate VALU ops.
__device__ __forceinline__ unsigned int pkbf(float lo, float hi) {
  unsigned int ul = __float_as_uint(lo) + 0x8000u;
  unsigned int uh = __float_as_uint(hi) + 0x8000u;
  return __builtin_amdgcn_perm(uh, ul, 0x07060302u);  // v_perm_b32
}

__device__ __forceinline__ unsigned short f2bf(float f) {
  return (unsigned short)((__float_as_uint(f) + 0x8000u) >> 16);
}

__device__ __forceinline__ v8f wmma_bf16(v16bf a, v16bf b, v8f c) {
  return __builtin_amdgcn_wmma_f32_16x16x32_bf16(false, a, false, b, (short)0, c,
                                                 false, false);
}

__device__ __forceinline__ v8f vzero8() {
  v8f z;
#pragma unroll
  for (int i = 0; i < 8; i++) z[i] = 0.0f;
  return z;
}

// Load one K-chunk's raw f32 data (16 values / lane) without converting.
template <bool NT>
__device__ __forceinline__ void load_chunk(const float* __restrict__ row, int off,
                                           f4* d) {
  const f4* p0 = (const f4*)(row + off);
  const f4* p1 = (const f4*)(row + off + 16);
  if (NT) {
    d[0] = __builtin_nontemporal_load(p0);
    d[1] = __builtin_nontemporal_load(p0 + 1);
    d[2] = __builtin_nontemporal_load(p1);
    d[3] = __builtin_nontemporal_load(p1 + 1);
  } else {
    d[0] = p0[0]; d[1] = p0[1]; d[2] = p1[0]; d[3] = p1[1];
  }
}

// Convert a raw chunk to an A fragment and run the 8 WMMAs of K-chunk kc.
__device__ __forceinline__ void mma_chunk(const f4* d, int kc,
                                          const unsigned short* __restrict__ sW,
                                          int lane, v8f* acc) {
  AFrag a;
#pragma unroll
  for (int q = 0; q < 4; q++) {
    a.ui[2 * q]     = pkbf(d[q].x, d[q].y);
    a.ui[2 * q + 1] = pkbf(d[q].z, d[q].w);
  }
#pragma unroll
  for (int nt = 0; nt < 8; nt++) {
    v16bf b = *(const v16bf*)(sW + (kc * 8 + nt) * 512 + lane * 16);
    acc[nt] = wmma_bf16(a.v, b, acc[nt]);
  }
}

// ---------------------------------------------------------------------------
__global__ void zero_f32(float* __restrict__ p, int n) {
  int i = blockIdx.x * blockDim.x + threadIdx.x;
  if (i < n) p[i] = 0.0f;
}

// ---------------------------------------------------------------------------
// Edge MLP: per wave, a 16-edge x 128-col tile of messages.
// W_edge (384x128 f32) staged once per block as bf16 in LDS, pre-swizzled into
// B-fragment layout: block (kc,nt) -> 32 lanes x 16 halfwords (32B/lane).
// ---------------------------------------------------------------------------
__global__ void edge_mlp_kernel(const float* __restrict__ nodes,
                                const float* __restrict__ edges,
                                const float* __restrict__ W_edge,
                                const float* __restrict__ b_edge,
                                const int*   __restrict__ senders,
                                const int*   __restrict__ receivers,
                                float* __restrict__ messages,
                                float* __restrict__ recv) {
  extern __shared__ unsigned short sW[];   // 96 KB: 96 blocks * 512 halfwords
  const int tid = threadIdx.x;

  // Stage W_edge -> bf16 fragment order.
  for (int idx = tid; idx < 384 * 128; idx += blockDim.x) {
    int k = idx >> 7;            // 0..383  (cin)
    int n = idx & 127;           // 0..127  (cout)
    int kc = k >> 5, kr = k & 31;
    int nt = n >> 4, nl = n & 15;
    int lane = nl + (((kr >> 3) & 1) << 4);
    int j = (kr & 7) + ((kr >> 4) << 3);
    sW[(kc * 8 + nt) * 512 + lane * 16 + j] = f2bf(W_edge[idx]);
  }
  __syncthreads();

  const int wave  = tid >> 5;
  const int lane  = tid & 31;
  const int m     = lane & 15;
  const bool hif  = lane >= 16;
  const int khalf = hif ? 8 : 0;
  const int mbase = hif ? 8 : 0;
  const int wavesPerGrid = gridDim.x * (blockDim.x >> 5);

  // Per-lane output-column biases: constant across tiles, hoisted.
  float bias[8];
#pragma unroll
  for (int nt = 0; nt < 8; nt++) bias[nt] = b_edge[nt * 16 + m];

  for (int tile = blockIdx.x * (blockDim.x >> 5) + wave; tile < NEDGES / 16;
       tile += wavesPerGrid) {
    const int e = tile * 16 + m;
    const int s = senders[e];
    const int r = receivers[e];
    const float* rowS = nodes + (size_t)s * DIM;
    const float* rowR = nodes + (size_t)r * DIM;
    const float* rowE = edges + (size_t)e * DIM;

    v8f acc[8];
#pragma unroll
    for (int t = 0; t < 8; t++) acc[t] = vzero8();

    // Software pipeline: fetch chunk kc+1 while chunk kc converts + MMAs.
    // K chunks 0..3: sender; 4..7: receiver; 8..11: edge features (NT).
    f4 cur[4], nxt[4];
    load_chunk<false>(rowS, khalf, cur);
#pragma unroll
    for (int kc = 0; kc < 12; kc++) {
      if (kc < 11) {
        const int kn = kc + 1;
        const float* rw = (kn < 4) ? rowS : (kn < 8) ? rowR : rowE;
        const int off = (kn & 3) * 32 + khalf;
        if (kn >= 8) load_chunk<true>(rw, off, nxt);
        else         load_chunk<false>(rw, off, nxt);
      }
      mma_chunk(cur, kc, sW, lane, acc);
#pragma unroll
      for (int q = 0; q < 4; q++) cur[q] = nxt[q];
    }

    // Epilogue: bias + ReLU, write messages (NT), scatter-add into recv.
    // Per-row base pointers; (nt) offsets fold into instruction immediates.
#pragma unroll
    for (int i = 0; i < 8; i++) {
      const int erow = tile * 16 + mbase + i;
      const int rcv  = receivers[erow];
      float* mp = messages + (size_t)erow * DIM + m;
      float* rp = recv + (size_t)rcv * DIM + m;
#pragma unroll
      for (int nt = 0; nt < 8; nt++) {
        float v = fmaxf(acc[nt][i] + bias[nt], 0.0f);
        __builtin_nontemporal_store(v, mp + nt * 16);
        atomicAdd(rp + nt * 16, v);
      }
    }
  }
}

// ---------------------------------------------------------------------------
// GRU: gi = recv@Wi + bi ; gh = nodes@Wh ; gates ; new_nodes.
// Wi and Wh (each 128x384) staged as bf16 fragments: 2 x 96 KB = 192 KB LDS.
// Each wave owns a 16-node x 16-gate-col block (6 accumulators).
// ---------------------------------------------------------------------------
__global__ void gru_kernel(const float* __restrict__ nodes,
                           const float* __restrict__ recv,
                           const float* __restrict__ Wi,
                           const float* __restrict__ bi,
                           const float* __restrict__ Wh,
                           const float* __restrict__ bh_n,
                           float* __restrict__ out_nodes) {
  extern __shared__ unsigned short sWib[];        // 192 KB
  unsigned short* sWi = sWib;
  unsigned short* sWh = sWib + 96 * 512;

  const int tid = threadIdx.x;
  for (int idx = tid; idx < 128 * 384; idx += blockDim.x) {
    int k = idx / 384;           // 0..127 (cin)
    int n = idx - k * 384;       // 0..383 (3*D gate cols)
    int kc = k >> 5, kr = k & 31;
    int nt = n >> 4, nl = n & 15;
    int lane = nl + (((kr >> 3) & 1) << 4);
    int j = (kr & 7) + ((kr >> 4) << 3);
    int dst = (kc * 24 + nt) * 512 + lane * 16 + j;
    sWi[dst] = f2bf(Wi[idx]);
    sWh[dst] = f2bf(Wh[idx]);
  }
  __syncthreads();

  const int wave  = tid >> 5;
  const int lane  = tid & 31;
  const int m     = lane & 15;
  const bool hif  = lane >= 16;
  const int khalf = hif ? 8 : 0;
  const int mbase = hif ? 8 : 0;
  const int wavesPerGrid = gridDim.x * (blockDim.x >> 5);
  const int nTasks = (NNODES / 16) * 8;     // 10000

  for (int task = blockIdx.x * (blockDim.x >> 5) + wave; task < nTasks;
       task += wavesPerGrid) {
    const int ntile = task >> 3;
    const int ct    = task & 7;             // which 16 of the 128 output cols
    const int row0  = ntile * 16;
    const float* rowRecv = recv  + (size_t)(row0 + m) * DIM;
    const float* rowNode = nodes + (size_t)(row0 + m) * DIM;

    v8f gir = vzero8(), giz = vzero8(), gin = vzero8();
    v8f ghr = vzero8(), ghz = vzero8(), ghn = vzero8();

#pragma unroll
    for (int kc = 0; kc < 4; kc++) {
      const int c0 = kc * 32 + khalf;
      f4 di[4], dh[4];
      load_chunk<false>(rowRecv, c0, di);
      load_chunk<false>(rowNode, c0, dh);
      AFrag ai, ah;
#pragma unroll
      for (int q = 0; q < 4; q++) {
        ai.ui[2 * q]     = pkbf(di[q].x, di[q].y);
        ai.ui[2 * q + 1] = pkbf(di[q].z, di[q].w);
        ah.ui[2 * q]     = pkbf(dh[q].x, dh[q].y);
        ah.ui[2 * q + 1] = pkbf(dh[q].z, dh[q].w);
      }

      v16bf br = *(const v16bf*)(sWi + (kc * 24 + ct     ) * 512 + lane * 16);
      v16bf bz = *(const v16bf*)(sWi + (kc * 24 + ct +  8) * 512 + lane * 16);
      v16bf bn = *(const v16bf*)(sWi + (kc * 24 + ct + 16) * 512 + lane * 16);
      gir = wmma_bf16(ai.v, br, gir);
      giz = wmma_bf16(ai.v, bz, giz);
      gin = wmma_bf16(ai.v, bn, gin);

      br = *(const v16bf*)(sWh + (kc * 24 + ct     ) * 512 + lane * 16);
      bz = *(const v16bf*)(sWh + (kc * 24 + ct +  8) * 512 + lane * 16);
      bn = *(const v16bf*)(sWh + (kc * 24 + ct + 16) * 512 + lane * 16);
      ghr = wmma_bf16(ah.v, br, ghr);
      ghz = wmma_bf16(ah.v, bz, ghz);
      ghn = wmma_bf16(ah.v, bn, ghn);
    }

    const int col = ct * 16 + m;            // 0..127
    const float br_ = bi[col];
    const float bz_ = bi[DIM + col];
    const float bn_ = bi[2 * DIM + col];
    const float bhn = bh_n[col];

#pragma unroll
    for (int i = 0; i < 8; i++) {
      const int row = row0 + mbase + i;
      const float ir  = gir[i] + br_;
      const float iz  = giz[i] + bz_;
      const float inn = gin[i] + bn_;
      const float rg  = 1.0f / (1.0f + __expf(-(ir + ghr[i])));
      const float zg  = 1.0f / (1.0f + __expf(-(iz + ghz[i])));
      const float ng  = tanhf(inn + rg * (ghn[i] + bhn));
      const float h   = nodes[(size_t)row * DIM + col];
      out_nodes[(size_t)row * DIM + col] = (1.0f - zg) * ng + zg * h;
    }
  }
}

// ---------------------------------------------------------------------------
extern "C" void kernel_launch(void* const* d_in, const int* in_sizes, int n_in,
                              void* d_out, int out_size, void* d_ws, size_t ws_size,
                              hipStream_t stream) {
  const float* nodes   = (const float*)d_in[0];
  const float* edges   = (const float*)d_in[1];
  const float* W_edge  = (const float*)d_in[2];
  const float* b_edge  = (const float*)d_in[3];
  const float* Wi      = (const float*)d_in[4];
  const float* bi      = (const float*)d_in[5];
  const float* Wh      = (const float*)d_in[6];
  const float* bh_n    = (const float*)d_in[7];
  const int*   senders = (const int*)d_in[8];
  const int*   recvrs  = (const int*)d_in[9];

  float* out_nodes = (float*)d_out;                       // [N, D]
  float* out_msgs  = out_nodes + (size_t)NNODES * DIM;    // [E, D]
  float* recvbuf   = (float*)d_ws;                        // [N, D] scratch

  const int nz = NNODES * DIM;
  zero_f32<<<(nz + 255) / 256, 256, 0, stream>>>(recvbuf, nz);

  // 96 KB LDS -> up to 3 blocks/WGP; 8 waves/block; grid-stride over 40k tiles.
  edge_mlp_kernel<<<1024, 256, 96 * 1024, stream>>>(
      nodes, edges, W_edge, b_edge, senders, recvrs, out_msgs, recvbuf);

  // 192 KB LDS (Wi+Wh resident); grid-stride over 10k wave tasks.
  gru_kernel<<<250, 256, 192 * 1024, stream>>>(
      nodes, recvbuf, Wi, bi, Wh, bh_n, out_nodes);
}
